// SNN_60567628808224
// MI455X (gfx1250) — matrix operand order repaired and checked by reading, not compiled, not run on hardware
//
#include <hip/hip_runtime.h>
#include <hip/hip_bf16.h>
#include <stdint.h>

// ---------------------------------------------------------------------------
// SNN forward, restructured for MI455X (gfx1250, wave32, WMMA):
//   1) pack_hi_lo   : f32 -> (bf16 hi, bf16 lo) splits of x, W1, W2
//   2) gemm1_wmma   : cur1 = x @ W1^T + b1   (3-term bf16 WMMA split)
//   3) spike_layer1 : elementwise 50-step LIF on cur1 -> h_sum + packed
//                     spike bits (u64 per (b,h), bit t = spk1[t])
//   4) gemm2_snn    : per (16b x 32o) tile: 50 GEMMs spk1[t] @ W2^T via
//                     bf16 WMMA (hi+lo split is EXACT since spikes are {0,1}),
//                     double-buffered async global->LDS staging, then the
//                     mem2 recurrence in LDS (two 25-step phases) -> out_sum
// ---------------------------------------------------------------------------

typedef __attribute__((ext_vector_type(16))) __bf16 v16bf;
typedef __attribute__((ext_vector_type(8)))  float  v8f;

#define SNN_BETA   0.9f
#define SNN_THRESH 1.0f
#define TSTEPS     50

#define NB   1024   // batch
#define DIN  1024
#define DH   2048
#define DOUT 256

// ---- CDNA5 async global->LDS copy (ASYNCcnt path, CDNA5 ISA) -----
__device__ __forceinline__ void async_ld_b128(void* lds, const void* g) {
  asm volatile("global_load_async_to_lds_b128 %0, %1, off"
               :: "v"((uint32_t)(uintptr_t)lds), "v"(g)
               : "memory");
}
__device__ __forceinline__ void s_wait_async0() {
  asm volatile("s_wait_asynccnt 0" ::: "memory");
}

// Assemble a v16bf WMMA fragment from two 16-byte chunks.
union FragU { uint4 q[2]; v16bf v; };
__device__ __forceinline__ v16bf load_frag(const __bf16* p0, const __bf16* p1) {
  FragU u;
  u.q[0] = *reinterpret_cast<const uint4*>(p0);
  u.q[1] = *reinterpret_cast<const uint4*>(p1);
  return u.v;
}

// ---------------------------------------------------------------------------
__global__ __launch_bounds__(256)
void pack_hi_lo(const float* __restrict__ src, __bf16* __restrict__ hi,
                __bf16* __restrict__ lo, int n) {
  int i = blockIdx.x * blockDim.x + threadIdx.x;
  if (i < n) {
    float v  = src[i];
    __bf16 h = (__bf16)v;
    hi[i] = h;
    lo[i] = (__bf16)(v - (float)h);
  }
}

// ---------------------------------------------------------------------------
// cur1[b,h] = sum_k x[b,k] * W1[h,k] + b1[h]
// A fragment (16x32 bf16): lane<16 holds K 0..15 of row m, lane>=16 K 16..31
// B fragment (32x16 bf16): lane<16 holds K 0..15 of col n, lane>=16 K 16..31
__global__ __launch_bounds__(256)
void gemm1_wmma(const __bf16* __restrict__ xh, const __bf16* __restrict__ xl,
                const __bf16* __restrict__ w1h, const __bf16* __restrict__ w1l,
                const float* __restrict__ b1, float* __restrict__ cur1) {
  const int lane = threadIdx.x & 31;
  const int wave = threadIdx.x >> 5;           // 0..7, one 16x16 n-subtile each
  const int b0   = blockIdx.x * 16;
  const int h0   = blockIdx.y * 128 + wave * 16;

  const int rowA  = b0 + (lane & 15);
  const int koffA = (lane < 16) ? 0 : 16;
  const int colB  = h0 + (lane & 15);
  const int koffB = (lane < 16) ? 0 : 16;

  v8f acc = {};
  for (int kk = 0; kk < DIN; kk += 32) {
    const __bf16* pah = xh  + (size_t)rowA * DIN + kk + koffA;
    const __bf16* pal = xl  + (size_t)rowA * DIN + kk + koffA;
    const __bf16* pbh = w1h + (size_t)colB * DIN + kk + koffB;
    const __bf16* pbl = w1l + (size_t)colB * DIN + kk + koffB;
    v16bf ah = load_frag(pah, pah + 8);
    v16bf al = load_frag(pal, pal + 8);
    v16bf bh = load_frag(pbh, pbh + 8);
    v16bf bl = load_frag(pbl, pbl + 8);
    acc = __builtin_amdgcn_wmma_f32_16x16x32_bf16(false, ah, false, bh,
                                                  (short)0, acc, false, false);
    acc = __builtin_amdgcn_wmma_f32_16x16x32_bf16(false, ah, false, bl,
                                                  (short)0, acc, false, false);
    acc = __builtin_amdgcn_wmma_f32_16x16x32_bf16(false, al, false, bh,
                                                  (short)0, acc, false, false);
  }

  // C layout: VGPR r, lanes 0-15 -> M=r, lanes 16-31 -> M=r+8; N = lane&15
  const int   n     = lane & 15;
  const int   mbase = (lane < 16) ? 0 : 8;
  const float bias  = b1[h0 + n];
#pragma unroll
  for (int r = 0; r < 8; ++r)
    cur1[(size_t)(b0 + mbase + r) * DH + h0 + n] = acc[r] + bias;
}

// ---------------------------------------------------------------------------
// Layer-1 LIF recurrence (cur1 constant over t): h_sum + 50 packed spike bits.
__global__ __launch_bounds__(256)
void spike_layer1(const float* __restrict__ cur1, float* __restrict__ hsum,
                  unsigned long long* __restrict__ bits) {
  int i = blockIdx.x * blockDim.x + threadIdx.x;
  if (i >= NB * DH) return;
  float c = cur1[i];
  float mem = 0.f, s = 0.f;
  unsigned long long m = 0ull;
#pragma unroll
  for (int t = 0; t < TSTEPS; ++t) {
    float reset = (mem > SNN_THRESH) ? SNN_THRESH : 0.f;  // reset on PREVIOUS mem
    mem = SNN_BETA * mem + c - reset;
    bool sp = (mem - SNN_THRESH) > 0.f;
    m |= ((unsigned long long)sp) << t;
    s += sp ? 1.f : 0.f;
  }
  hsum[i] = s;
  bits[i] = m;
}

// ---------------------------------------------------------------------------
// Fused layer-2.  WG = 512 threads = 16 waves, tile = 16 b x 32 o, all 50 t.
// Wave w owns t = w, w+16, w+32, w+48 (<= 4 slots).  Each expanded spike
// A-tile feeds 4 WMMAs (2 o-subtiles x hi/lo) so bit->bf16 VALU expansion
// (~56 ops) is balanced against ~64 matrix cycles.  Staging is double-
// buffered: async copies of chunk i+1 overlap WMMA work on chunk i.
#define G2_STAGE 8192   // per-buffer: 4 KB spike words + 2 KB W2hi + 2 KB W2lo

__device__ __forceinline__ void stage_tiles(char* sb,
    const unsigned long long* __restrict__ bits,
    const __bf16* __restrict__ w2h, const __bf16* __restrict__ w2l,
    int tid, int b0, int o0, int kk) {
  unsigned long long* sbits = (unsigned long long*)sb;          // 512 u64
  __bf16* swh = (__bf16*)(sb + 4096);                           // 32o x 32k
  __bf16* swl = (__bf16*)(sb + 6144);                           // 32o x 32k
  if (tid < 256) {        // 16b x 32h spike words, 16 B per thread
    int e = tid * 2, m = e >> 5, k = e & 31;
    async_ld_b128(sbits + e, bits + (size_t)(b0 + m) * DH + kk + k);
  } else if (tid < 384) { // W2 hi tile, 16 B per thread
    int idx = (tid - 256) * 8, o = idx >> 5, k = idx & 31;
    async_ld_b128(swh + idx, w2h + (size_t)(o0 + o) * DH + kk + k);
  } else {                // W2 lo tile
    int idx = (tid - 384) * 8, o = idx >> 5, k = idx & 31;
    async_ld_b128(swl + idx, w2l + (size_t)(o0 + o) * DH + kk + k);
  }
}

__global__ __launch_bounds__(512)
void gemm2_snn(const unsigned long long* __restrict__ bits,
               const __bf16* __restrict__ w2h, const __bf16* __restrict__ w2l,
               const float* __restrict__ b2, float* __restrict__ out2) {
  // 25 t-steps x 512 elements x f32 = 50 KB; staging buffers live in the
  // first 16 KB during the K loop (recurrence region reused afterwards).
  __shared__ __align__(16) char smem[25 * 512 * 4];

  const int tid  = threadIdx.x;
  const int lane = tid & 31;
  const int wave = tid >> 5;                 // 0..15, owns t = wave + 16*s
  const int b0   = blockIdx.x * 16;
  const int o0   = blockIdx.y * 32;

  v8f acc[4][2] = {};                        // [t-slot][o-subtile]

  const int mrow  = lane & 15;
  const int koffA = (lane < 16) ? 0 : 16;
  const int colB  = lane & 15;
  const int koffB = (lane < 16) ? 0 : 16;

  stage_tiles(smem, bits, w2h, w2l, tid, b0, o0, 0);   // prologue: chunk 0

  for (int kk = 0, i = 0; kk < DH; kk += 32, ++i) {
    s_wait_async0();            // my async copies for chunk i have landed
    __syncthreads();            // everyone's copies landed; buf (i^1) free
    if (kk + 32 < DH)           // prefetch chunk i+1 into the other buffer
      stage_tiles(smem + ((i + 1) & 1) * G2_STAGE, bits, w2h, w2l,
                  tid, b0, o0, kk + 32);

    char* sb = smem + (i & 1) * G2_STAGE;
    unsigned long long* sbits = (unsigned long long*)sb;
    __bf16* swh = (__bf16*)(sb + 4096);
    __bf16* swl = (__bf16*)(sb + 6144);

    // --- B fragments from LDS: 2 o-subtiles x (hi, lo) ------------------
    const __bf16* p;
    p = swh + (colB)      * 32 + koffB;  v16bf bh0 = load_frag(p, p + 8);
    p = swh + (colB + 16) * 32 + koffB;  v16bf bh1 = load_frag(p, p + 8);
    p = swl + (colB)      * 32 + koffB;  v16bf bl0 = load_frag(p, p + 8);
    p = swl + (colB + 16) * 32 + koffB;  v16bf bl1 = load_frag(p, p + 8);

    // --- this lane's 16 spike words (reused across all t slots) ---------
    unsigned long long w[16];
    {
      const ulonglong2* p0 = (const ulonglong2*)(sbits + mrow * 32 + koffA);
      const ulonglong2* p1 = (const ulonglong2*)(sbits + mrow * 32 + 8 + koffA);
#pragma unroll
      for (int q = 0; q < 4; ++q) {
        ulonglong2 a = p0[q]; w[2 * q]     = a.x; w[2 * q + 1]     = a.y;
        ulonglong2 b = p1[q]; w[8 + 2 * q] = b.x; w[8 + 2 * q + 1] = b.y;
      }
    }

    // --- expand bits -> bf16 A, 4 WMMAs per t slot ----------------------
#pragma unroll
    for (int s = 0; s < 4; ++s) {
      int t = wave + 16 * s;               // wave-uniform: EXEC stays all-1s
      if (t < TSTEPS) {
        union { uint32_t d[8]; v16bf v; } au;
#pragma unroll
        for (int j = 0; j < 8; ++j) {
          uint32_t lo = ((w[2 * j]     >> t) & 1ull) ? 0x00003F80u : 0u;
          uint32_t hi = ((w[2 * j + 1] >> t) & 1ull) ? 0x3F800000u : 0u;
          au.d[j] = lo | hi;
        }
        acc[s][0] = __builtin_amdgcn_wmma_f32_16x16x32_bf16(
            false, au.v, false, bh0, (short)0, acc[s][0], false, false);
        acc[s][0] = __builtin_amdgcn_wmma_f32_16x16x32_bf16(
            false, au.v, false, bl0, (short)0, acc[s][0], false, false);
        acc[s][1] = __builtin_amdgcn_wmma_f32_16x16x32_bf16(
            false, au.v, false, bh1, (short)0, acc[s][1], false, false);
        acc[s][1] = __builtin_amdgcn_wmma_f32_16x16x32_bf16(
            false, au.v, false, bl1, (short)0, acc[s][1], false, false);
      }
    }
  }

  // --- mem2 recurrence in two 25-step phases (LDS reused, 50 KB) --------
  float* scur = (float*)smem;
  const int n     = lane & 15;
  const int mbase = (lane < 16) ? 0 : 8;
  const float bias = b2[o0 + (tid & 31)];
  float mem = 0.f, osum = 0.f;

#pragma unroll
  for (int phase = 0; phase < 2; ++phase) {
    __syncthreads();
    // dump cur2[t] tiles for t in [phase*25, phase*25+25)
#pragma unroll
    for (int s = 0; s < 4; ++s) {
      int t = wave + 16 * s;
      if (t >= phase * 25 && t < phase * 25 + 25) {
        int lt = t - phase * 25;
#pragma unroll
        for (int sub = 0; sub < 2; ++sub)
#pragma unroll
          for (int r = 0; r < 8; ++r)
            scur[lt * 512 + (mbase + r) * 32 + sub * 16 + n] = acc[s][sub][r];
      }
    }
    __syncthreads();
    // one thread per (b, o) element; carry mem/osum across phases
#pragma unroll
    for (int lt = 0; lt < 25; ++lt) {
      float cur   = scur[lt * 512 + tid] + bias;
      float reset = (mem > SNN_THRESH) ? SNN_THRESH : 0.f;
      mem = SNN_BETA * mem + cur - reset;
      osum += ((mem - SNN_THRESH) > 0.f) ? 1.f : 0.f;
    }
  }
  out2[(size_t)(b0 + (tid >> 5)) * DOUT + o0 + (tid & 31)] = osum;
}

// ---------------------------------------------------------------------------
extern "C" void kernel_launch(void* const* d_in, const int* in_sizes, int n_in,
                              void* d_out, int out_size, void* d_ws, size_t ws_size,
                              hipStream_t stream) {
  const float* x  = (const float*)d_in[0];  // [1024,1024]
  const float* W1 = (const float*)d_in[1];  // [2048,1024]
  const float* b1 = (const float*)d_in[2];  // [2048]
  const float* W2 = (const float*)d_in[3];  // [256,2048]
  const float* b2 = (const float*)d_in[4];  // [256]
  float* out = (float*)d_out;               // h_sum [1024,2048] ++ out_sum [1024,256]

  // workspace layout (38 MB total)
  char* w = (char*)d_ws;
  float*              cur1 = (float*)(w);                               // 8 MB
  unsigned long long* bits = (unsigned long long*)(w + (8u  << 20));    // 16 MB
  __bf16* xh  = (__bf16*)(w + (24u << 20));                             // 2 MB
  __bf16* xl  = (__bf16*)(w + (26u << 20));                             // 2 MB
  __bf16* w1h = (__bf16*)(w + (28u << 20));                             // 4 MB
  __bf16* w1l = (__bf16*)(w + (32u << 20));                             // 4 MB
  __bf16* w2h = (__bf16*)(w + (36u << 20));                             // 1 MB
  __bf16* w2l = (__bf16*)(w + (37u << 20));                             // 1 MB

  // 1) hi/lo bf16 splits
  pack_hi_lo<<<(NB * DIN) / 256, 256, 0, stream>>>(x,  xh,  xl,  NB * DIN);
  pack_hi_lo<<<(DH * DIN) / 256, 256, 0, stream>>>(W1, w1h, w1l, DH * DIN);
  pack_hi_lo<<<(DOUT * DH) / 256, 256, 0, stream>>>(W2, w2h, w2l, DOUT * DH);

  // 2) cur1 = x @ W1^T + b1
  gemm1_wmma<<<dim3(NB / 16, DH / 128), 256, 0, stream>>>(xh, xl, w1h, w1l, b1, cur1);

  // 3) layer-1 LIF: h_sum + packed spike bits
  spike_layer1<<<(NB * DH) / 256, 256, 0, stream>>>(cur1, out, bits);

  // 4) fused layer-2: 50x (spk1 @ W2^T) + mem2 recurrence -> out_sum
  gemm2_snn<<<dim3(NB / 16, DOUT / 32), 512, 0, stream>>>(
      bits, w2h, w2l, b2, out + (size_t)NB * DH);
}
